// DualEdgeGraphConvBlock_23089744183608
// MI455X (gfx1250) — matrix operand order repaired and checked by reading, not compiled, not run on hardware
//
#include <hip/hip_runtime.h>

// ---------------------------------------------------------------------------
// DualEdgeGraphConvBlock for MI455X (gfx1250, wave32, WMMA bf16 16x16x32).
// Pipeline:
//   prep_x / prep_w      : fp32 -> bf16 copies, row norms, zero BN stats
//   knn_x_kernel         : WMMA-based 64-d pairwise dots -> top-8 (diag=INF)
//   knn_pos_kernel       : 3-d distances -> top-32 (writes idx2 to d_out)
//   conv1_stats/passB    : y1 = f1 @ W1^T (WMMA), BN stats, affine+LReLU+max_k -> h1 (bf16)
//   conv2_stats          : y2 = f2 @ W2^T (WMMA), BN stats
//   conv3_stats          : recompute h2 (LDS bf16), y3 = h2 @ W3^T (WMMA), BN stats
//   conv3_final          : recompute h2,h3; max over k=32 -> out (f32)
// ---------------------------------------------------------------------------

constexpr int B_   = 2;
constexpr int N_   = 4096;
constexpr int C_   = 64;
constexpr int C1_  = 128;
constexpr int HID_ = 128;
constexpr int C2_  = 256;
constexpr int OC_  = 64;
constexpr int KX_  = 8;
constexpr int K2_  = 32;
constexpr int K1T_ = 14;
constexpr float SLOPE_ = 0.2f;
constexpr float EPS_   = 1e-5f;

#define FINF __builtin_inff()

typedef __attribute__((ext_vector_type(16))) __bf16 bf16x16;
typedef __attribute__((ext_vector_type(8)))  __bf16 bf16x8;
typedef __attribute__((ext_vector_type(8)))  float  f32x8;

__device__ inline f32x8 wmma_bf16(bf16x16 a, bf16x16 b, f32x8 c) {
  return __builtin_amdgcn_wmma_f32_16x16x32_bf16(false, a, false, b, (short)0, c,
                                                 false, false);
}

__device__ inline bf16x16 pack16(bf16x8 lo, bf16x8 hi) {
  bf16x16 r;
#pragma unroll
  for (int i = 0; i < 8; ++i) { r[i] = lo[i]; r[i + 8] = hi[i]; }
  return r;
}

// A-operand (16x32 bf16, MxK): lane L holds row M=L&15; K chunks at
// kBase + half*16 and kBase + half*16 + 8 (half = L>>4), each 8 contiguous bf16.
__device__ inline bf16x16 loadA_row(const __bf16* row, int kBase, int half) {
  bf16x8 c0 = *(const bf16x8*)(row + kBase + half * 16);
  bf16x8 c1 = *(const bf16x8*)(row + kBase + half * 16 + 8);
  return pack16(c0, c1);
}

// B-operand (32x16 bf16, KxN) built from W[o][c] (row major): lane L holds
// column N (=output o = oBase + (L&15)), K = kBase + (L>>4)*16 + e.
__device__ inline bf16x16 loadB_W(const __bf16* W, int Cin, int oBase, int kBase,
                                  int lane) {
  const __bf16* p = W + (size_t)(oBase + (lane & 15)) * Cin + kBase + (lane >> 4) * 16;
  return pack16(*(const bf16x8*)(p), *(const bf16x8*)(p + 8));
}

// Edge feature chunk: c < chalf -> nbr[c]-ctr[c]; else ctr[c-chalf].
__device__ inline bf16x8 edge_chunk(const __bf16* ctr, const __bf16* nbr, int off,
                                    int chalf) {
  bf16x8 r;
  if (off < chalf) {
    bf16x8 a = *(const bf16x8*)(nbr + off);
    bf16x8 b = *(const bf16x8*)(ctr + off);
#pragma unroll
    for (int i = 0; i < 8; ++i) r[i] = (__bf16)((float)a[i] - (float)b[i]);
  } else {
    r = *(const bf16x8*)(ctr + (off - chalf));
  }
  return r;
}

__device__ inline bf16x16 loadA_edge(const __bf16* ctr, const __bf16* nbr, int kBase,
                                     int half, int chalf) {
  return pack16(edge_chunk(ctr, nbr, kBase + half * 16, chalf),
                edge_chunk(ctr, nbr, kBase + half * 16 + 8, chalf));
}

template <int KK>
__device__ inline void insert_sorted(float (&kd)[KK], int (&ki)[KK], float d, int j) {
  if (d >= kd[KK - 1]) return;
#pragma unroll
  for (int t = KK - 1; t > 0; --t) {
    if (d >= kd[t - 1]) { kd[t] = d; ki[t] = j; return; }
    kd[t] = kd[t - 1]; ki[t] = ki[t - 1];
  }
  kd[0] = d; ki[0] = j;
}

// ---------------------------------------------------------------------------
__global__ void prep_x(const float* __restrict__ x, __bf16* __restrict__ xbf,
                       float* __restrict__ xnorm) {
  const int t = blockIdx.x * blockDim.x + threadIdx.x;
  if (t >= B_ * N_) return;
  float nrm = 0.f;
#pragma unroll 8
  for (int c = 0; c < C_; ++c) {
    float v = x[(size_t)t * C_ + c];
    nrm += v * v;
    xbf[(size_t)t * C_ + c] = (__bf16)v;
  }
  xnorm[t] = nrm;
}

__global__ void prep_w(const float* __restrict__ W1, const float* __restrict__ W2,
                       const float* __restrict__ W3, __bf16* __restrict__ W1b,
                       __bf16* __restrict__ W2b, __bf16* __restrict__ W3b,
                       float* __restrict__ stats) {
  const int t = blockIdx.x * blockDim.x + threadIdx.x;
  if (t < 16384) W1b[t] = (__bf16)W1[t];
  else if (t < 49152) W2b[t - 16384] = (__bf16)W2[t - 16384];
  else if (t < 57344) W3b[t - 49152] = (__bf16)W3[t - 49152];
  if (t < 1280) stats[t] = 0.f;
}

// ---------------------------------------------------------------------------
// KNN over 64-d features via WMMA dot tiles. Block: 256 thr (8 waves), 32 rows.
__global__ __launch_bounds__(256) void knn_x_kernel(const __bf16* __restrict__ xbf,
                                                    const float* __restrict__ xnorm,
                                                    int* __restrict__ idx1x) {
  const int b = blockIdx.y;
  const int r0 = blockIdx.x * 32;
  const int lane = threadIdx.x & 31;
  const int wave = threadIdx.x >> 5;
  const int half = lane >> 4;
  const __bf16* xb = xbf + (size_t)b * N_ * C_;
  const float* xn = xnorm + (size_t)b * N_;

  __shared__ float dbuf[32 * 128];
  __shared__ float mD[32 * 64];
  __shared__ int   mI[32 * 64];

  float kd[8]; int ki[8];
#pragma unroll
  for (int i = 0; i < 8; ++i) { kd[i] = FINF; ki[i] = 0; }

  const int lr = threadIdx.x >> 3;  // selection row 0..31
  const int ls = threadIdx.x & 7;   // selection subset 0..7

  for (int ct = 0; ct < N_ / 128; ++ct) {
    const int cBase = ct * 128;
#pragma unroll
    for (int rh = 0; rh < 2; ++rh) {
      f32x8 acc = {};
      const int mrow = r0 + rh * 16 + (lane & 15);
      const int col = cBase + wave * 16 + (lane & 15);
#pragma unroll
      for (int j = 0; j < 2; ++j) {
        bf16x16 a = loadA_row(xb + (size_t)mrow * C_, j * 32, half);
        const __bf16* pB = xb + (size_t)col * C_ + j * 32 + half * 16;
        bf16x16 bm = pack16(*(const bf16x8*)pB, *(const bf16x8*)(pB + 8));
        acc = wmma_bf16(a, bm, acc);
      }
      const float cn = xn[col];
#pragma unroll
      for (int r = 0; r < 8; ++r) {
        const int row = r0 + rh * 16 + half * 8 + r;
        float d = xn[row] + cn - 2.0f * acc[r];
        if (row == col) d = FINF;
        dbuf[(rh * 16 + half * 8 + r) * 128 + wave * 16 + (lane & 15)] = d;
      }
    }
    __syncthreads();
#pragma unroll
    for (int c = 0; c < 16; ++c) {
      const int lc = ls * 16 + c;
      insert_sorted<8>(kd, ki, dbuf[lr * 128 + lc], cBase + lc);
    }
    __syncthreads();
  }
#pragma unroll
  for (int i = 0; i < 8; ++i) { mD[lr * 64 + ls * 8 + i] = kd[i]; mI[lr * 64 + ls * 8 + i] = ki[i]; }
  __syncthreads();
  if (ls == 0) {
    float fd[8]; int fi[8];
#pragma unroll
    for (int i = 0; i < 8; ++i) { fd[i] = FINF; fi[i] = 0; }
    for (int c = 0; c < 64; ++c) insert_sorted<8>(fd, fi, mD[lr * 64 + c], mI[lr * 64 + c]);
#pragma unroll
    for (int i = 0; i < 8; ++i) idx1x[((size_t)b * N_ + r0 + lr) * KX_ + i] = fi[i];
  }
}

// KNN over 3-d positions. Block: 128 thr, 32 rows x 4 scanners.
__global__ __launch_bounds__(128) void knn_pos_kernel(const float* __restrict__ pos,
                                                      int* __restrict__ idx2out) {
  const int b = blockIdx.y;
  const int r0 = blockIdx.x * 32;
  const int lr = threadIdx.x >> 2;
  const int ls = threadIdx.x & 3;
  const float* pp = pos + (size_t)b * N_ * 3;
  const int row = r0 + lr;
  const float px = pp[row * 3 + 0], py = pp[row * 3 + 1], pz = pp[row * 3 + 2];

  float kd[32]; int ki[32];
#pragma unroll
  for (int i = 0; i < 32; ++i) { kd[i] = FINF; ki[i] = 0; }

  for (int j = ls; j < N_; j += 4) {
    if (j == row) continue;
    const float dx = pp[j * 3 + 0] - px;
    const float dy = pp[j * 3 + 1] - py;
    const float dz = pp[j * 3 + 2] - pz;
    insert_sorted<32>(kd, ki, dx * dx + dy * dy + dz * dz, j);
  }
  __shared__ float mD[32 * 128];
  __shared__ int   mI[32 * 128];
#pragma unroll
  for (int i = 0; i < 32; ++i) { mD[lr * 128 + ls * 32 + i] = kd[i]; mI[lr * 128 + ls * 32 + i] = ki[i]; }
  __syncthreads();
  if (ls == 0) {
    float fd[32]; int fi[32];
#pragma unroll
    for (int i = 0; i < 32; ++i) { fd[i] = FINF; fi[i] = 0; }
    for (int c = 0; c < 128; ++c) insert_sorted<32>(fd, fi, mD[lr * 128 + c], mI[lr * 128 + c]);
#pragma unroll
    for (int i = 0; i < 32; ++i) idx2out[((size_t)b * N_ + row) * K2_ + i] = fi[i];
  }
}

// ---------------------------------------------------------------------------
// Conv1 pass A: BN stats of y1 = f1 @ W1^T. Block: 128 samples x 128 outputs.
__global__ __launch_bounds__(256) void conv1_stats(const __bf16* __restrict__ xbf,
    const __bf16* __restrict__ W1b, const int* __restrict__ idx1x,
    const int* __restrict__ idx2, float* __restrict__ gsum, float* __restrict__ gsq) {
  const int lane = threadIdx.x & 31, wave = threadIdx.x >> 5, half = lane >> 4;
  const int sBase = blockIdx.x * 128;
  bf16x16 Bt[4];
#pragma unroll
  for (int j = 0; j < 4; ++j) Bt[j] = loadB_W(W1b, C1_, wave * 16, j * 32, lane);
  float lsum = 0.f, lsq = 0.f;
  for (int rt = 0; rt < 8; ++rt) {
    const int s = sBase + rt * 16 + (lane & 15);
    const int b = s / (N_ * K1T_);
    const int rem = s - b * (N_ * K1T_);
    const int n = rem / K1T_;
    const int k = rem - n * K1T_;
    const __bf16* ctr = xbf + (size_t)(b * N_ + n) * C_;
    const int nb = (k < KX_) ? idx1x[(size_t)(b * N_ + n) * KX_ + k]
                             : idx2[(size_t)(b * N_ + n) * K2_ + (k - KX_)];
    const __bf16* nbr = xbf + (size_t)(b * N_ + nb) * C_;
    f32x8 acc = {};
#pragma unroll
    for (int j = 0; j < 4; ++j)
      acc = wmma_bf16(loadA_edge(ctr, nbr, j * 32, half, C_), Bt[j], acc);
#pragma unroll
    for (int r = 0; r < 8; ++r) { lsum += acc[r]; lsq += acc[r] * acc[r]; }
  }
  __shared__ float bs[128], bq[128];
  if (threadIdx.x < 128) { bs[threadIdx.x] = 0.f; bq[threadIdx.x] = 0.f; }
  __syncthreads();
  atomicAdd(&bs[wave * 16 + (lane & 15)], lsum);
  atomicAdd(&bq[wave * 16 + (lane & 15)], lsq);
  __syncthreads();
  if (threadIdx.x < 128) {
    atomicAdd(&gsum[threadIdx.x], bs[threadIdx.x]);
    atomicAdd(&gsq[threadIdx.x], bq[threadIdx.x]);
  }
}

__global__ void finalize_bn(const float* __restrict__ sum, const float* __restrict__ sq,
                            const float* __restrict__ g, const float* __restrict__ bb,
                            float* __restrict__ scale, float* __restrict__ shift,
                            int nch, float invS) {
  const int o = threadIdx.x;
  if (o >= nch) return;
  const float mu = sum[o] * invS;
  const float var = sq[o] * invS - mu * mu;
  const float sc = g[o] * rsqrtf(var + EPS_);
  scale[o] = sc;
  shift[o] = bb[o] - mu * sc;
}

// Conv1 pass B: recompute y1, affine+LReLU, max over k=14 -> h1 (bf16).
// Block handles 8 points (112 samples = 7 row tiles) x 128 outputs.
__global__ __launch_bounds__(256) void conv1_passB(const __bf16* __restrict__ xbf,
    const __bf16* __restrict__ W1b, const int* __restrict__ idx1x,
    const int* __restrict__ idx2, const float* __restrict__ sc1,
    const float* __restrict__ sh1, __bf16* __restrict__ h1b) {
  __shared__ float ybuf[112 * 128];
  const int lane = threadIdx.x & 31, wave = threadIdx.x >> 5, half = lane >> 4;
  const int b = blockIdx.y;
  const int nBase = blockIdx.x * 8;
  bf16x16 Bt[4];
#pragma unroll
  for (int j = 0; j < 4; ++j) Bt[j] = loadB_W(W1b, C1_, wave * 16, j * 32, lane);
  const int oc = wave * 16 + (lane & 15);
  const float s1 = sc1[oc], bb1 = sh1[oc];
  for (int rt = 0; rt < 7; ++rt) {
    const int sl = rt * 16 + (lane & 15);
    const int nl = sl / 14;
    const int k = sl - nl * 14;
    const int n = nBase + nl;
    const __bf16* ctr = xbf + (size_t)(b * N_ + n) * C_;
    const int nb = (k < KX_) ? idx1x[(size_t)(b * N_ + n) * KX_ + k]
                             : idx2[(size_t)(b * N_ + n) * K2_ + (k - KX_)];
    const __bf16* nbr = xbf + (size_t)(b * N_ + nb) * C_;
    f32x8 acc = {};
#pragma unroll
    for (int j = 0; j < 4; ++j)
      acc = wmma_bf16(loadA_edge(ctr, nbr, j * 32, half, C_), Bt[j], acc);
#pragma unroll
    for (int r = 0; r < 8; ++r) {
      float y = acc[r] * s1 + bb1;
      y = y > 0.f ? y : SLOPE_ * y;
      ybuf[(rt * 16 + half * 8 + r) * 128 + oc] = y;
    }
  }
  __syncthreads();
#pragma unroll
  for (int i = 0; i < 4; ++i) {
    const int idx = threadIdx.x + i * 256;
    const int nl = idx >> 7, o = idx & 127;
    float m = -FINF;
#pragma unroll
    for (int k = 0; k < 14; ++k) m = fmaxf(m, ybuf[(nl * 14 + k) * 128 + o]);
    h1b[((size_t)(b * N_) + nBase + nl) * HID_ + o] = (__bf16)m;
  }
}

// Conv2 pass A: BN stats of y2 = f2 @ W2^T (K=256). 128 samples x 128 outputs.
__global__ __launch_bounds__(256) void conv2_stats(const __bf16* __restrict__ h1b,
    const __bf16* __restrict__ W2b, const int* __restrict__ idx2,
    float* __restrict__ gsum, float* __restrict__ gsq) {
  const int lane = threadIdx.x & 31, wave = threadIdx.x >> 5, half = lane >> 4;
  const int sBase = blockIdx.x * 128;
  bf16x16 Bt[8];
#pragma unroll
  for (int j = 0; j < 8; ++j) Bt[j] = loadB_W(W2b, C2_, wave * 16, j * 32, lane);
  float lsum = 0.f, lsq = 0.f;
  for (int rt = 0; rt < 8; ++rt) {
    const int s = sBase + rt * 16 + (lane & 15);
    const int b = s >> 17;
    const int rem = s & (N_ * K2_ - 1);
    const int n = rem >> 5, k = rem & 31;
    const __bf16* ctr = h1b + (size_t)(b * N_ + n) * HID_;
    const int nb = idx2[(size_t)(b * N_ + n) * K2_ + k];
    const __bf16* nbr = h1b + (size_t)(b * N_ + nb) * HID_;
    f32x8 acc = {};
#pragma unroll
    for (int j = 0; j < 8; ++j)
      acc = wmma_bf16(loadA_edge(ctr, nbr, j * 32, half, HID_), Bt[j], acc);
#pragma unroll
    for (int r = 0; r < 8; ++r) { lsum += acc[r]; lsq += acc[r] * acc[r]; }
  }
  __shared__ float bs[128], bq[128];
  if (threadIdx.x < 128) { bs[threadIdx.x] = 0.f; bq[threadIdx.x] = 0.f; }
  __syncthreads();
  atomicAdd(&bs[wave * 16 + (lane & 15)], lsum);
  atomicAdd(&bq[wave * 16 + (lane & 15)], lsq);
  __syncthreads();
  if (threadIdx.x < 128) {
    atomicAdd(&gsum[threadIdx.x], bs[threadIdx.x]);
    atomicAdd(&gsq[threadIdx.x], bq[threadIdx.x]);
  }
}

// Conv3 pass A: recompute h2 (LDS, bf16), y3 = h2 @ W3^T, BN stats (64 ch).
__global__ __launch_bounds__(256) void conv3_stats(const __bf16* __restrict__ h1b,
    const __bf16* __restrict__ W2b, const __bf16* __restrict__ W3b,
    const int* __restrict__ idx2, const float* __restrict__ sc2,
    const float* __restrict__ sh2, float* __restrict__ gsum, float* __restrict__ gsq) {
  __shared__ __bf16 h2buf[128 * 128];
  __shared__ float bs[64], bq[64];
  const int lane = threadIdx.x & 31, wave = threadIdx.x >> 5, half = lane >> 4;
  const int sBase = blockIdx.x * 128;
  {
    bf16x16 Bt[8];
#pragma unroll
    for (int j = 0; j < 8; ++j) Bt[j] = loadB_W(W2b, C2_, wave * 16, j * 32, lane);
    const int oc = wave * 16 + (lane & 15);
    const float s2 = sc2[oc], b2 = sh2[oc];
    for (int rt = 0; rt < 8; ++rt) {
      const int s = sBase + rt * 16 + (lane & 15);
      const int b = s >> 17;
      const int rem = s & (N_ * K2_ - 1);
      const int n = rem >> 5, k = rem & 31;
      const __bf16* ctr = h1b + (size_t)(b * N_ + n) * HID_;
      const int nb = idx2[(size_t)(b * N_ + n) * K2_ + k];
      const __bf16* nbr = h1b + (size_t)(b * N_ + nb) * HID_;
      f32x8 acc = {};
#pragma unroll
      for (int j = 0; j < 8; ++j)
        acc = wmma_bf16(loadA_edge(ctr, nbr, j * 32, half, HID_), Bt[j], acc);
#pragma unroll
      for (int r = 0; r < 8; ++r) {
        float y = acc[r] * s2 + b2;
        y = y > 0.f ? y : SLOPE_ * y;
        h2buf[(rt * 16 + half * 8 + r) * 128 + oc] = (__bf16)y;
      }
    }
  }
  __syncthreads();
  {
    const int ct = wave & 3, rg = (wave >> 2) * 4;
    bf16x16 Bt3[4];
#pragma unroll
    for (int j = 0; j < 4; ++j) Bt3[j] = loadB_W(W3b, HID_, ct * 16, j * 32, lane);
    float lsum = 0.f, lsq = 0.f;
    for (int rr = 0; rr < 4; ++rr) {
      const __bf16* row = &h2buf[(size_t)((rg + rr) * 16 + (lane & 15)) * 128];
      f32x8 acc = {};
#pragma unroll
      for (int j = 0; j < 4; ++j)
        acc = wmma_bf16(loadA_row(row, j * 32, half), Bt3[j], acc);
#pragma unroll
      for (int r = 0; r < 8; ++r) { lsum += acc[r]; lsq += acc[r] * acc[r]; }
    }
    if (threadIdx.x < 64) { bs[threadIdx.x] = 0.f; bq[threadIdx.x] = 0.f; }
    __syncthreads();
    atomicAdd(&bs[ct * 16 + (lane & 15)], lsum);
    atomicAdd(&bq[ct * 16 + (lane & 15)], lsq);
    __syncthreads();
    if (threadIdx.x < 64) {
      atomicAdd(&gsum[threadIdx.x], bs[threadIdx.x]);
      atomicAdd(&gsq[threadIdx.x], bq[threadIdx.x]);
    }
  }
}

// Conv3 pass B: recompute h2 -> h3, max over k=32 -> out.
__global__ __launch_bounds__(256) void conv3_final(const __bf16* __restrict__ h1b,
    const __bf16* __restrict__ W2b, const __bf16* __restrict__ W3b,
    const int* __restrict__ idx2, const float* __restrict__ sc2,
    const float* __restrict__ sh2, const float* __restrict__ sc3,
    const float* __restrict__ sh3, float* __restrict__ out) {
  __shared__ union SMem { __bf16 h2[128 * 128]; float h3[128 * 64]; } sm;
  const int lane = threadIdx.x & 31, wave = threadIdx.x >> 5, half = lane >> 4;
  const int sBase = blockIdx.x * 128;
  {
    bf16x16 Bt[8];
#pragma unroll
    for (int j = 0; j < 8; ++j) Bt[j] = loadB_W(W2b, C2_, wave * 16, j * 32, lane);
    const int oc = wave * 16 + (lane & 15);
    const float s2 = sc2[oc], b2 = sh2[oc];
    for (int rt = 0; rt < 8; ++rt) {
      const int s = sBase + rt * 16 + (lane & 15);
      const int b = s >> 17;
      const int rem = s & (N_ * K2_ - 1);
      const int n = rem >> 5, k = rem & 31;
      const __bf16* ctr = h1b + (size_t)(b * N_ + n) * HID_;
      const int nb = idx2[(size_t)(b * N_ + n) * K2_ + k];
      const __bf16* nbr = h1b + (size_t)(b * N_ + nb) * HID_;
      f32x8 acc = {};
#pragma unroll
      for (int j = 0; j < 8; ++j)
        acc = wmma_bf16(loadA_edge(ctr, nbr, j * 32, half, HID_), Bt[j], acc);
#pragma unroll
      for (int r = 0; r < 8; ++r) {
        float y = acc[r] * s2 + b2;
        y = y > 0.f ? y : SLOPE_ * y;
        sm.h2[(rt * 16 + half * 8 + r) * 128 + oc] = (__bf16)y;
      }
    }
  }
  __syncthreads();
  float h3v[4][8];
  const int ct = wave & 3, rg = (wave >> 2) * 4;
  {
    bf16x16 Bt3[4];
#pragma unroll
    for (int j = 0; j < 4; ++j) Bt3[j] = loadB_W(W3b, HID_, ct * 16, j * 32, lane);
    const int oc = ct * 16 + (lane & 15);
    const float s3 = sc3[oc], b3 = sh3[oc];
    for (int rr = 0; rr < 4; ++rr) {
      const __bf16* row = &sm.h2[(size_t)((rg + rr) * 16 + (lane & 15)) * 128];
      f32x8 acc = {};
#pragma unroll
      for (int j = 0; j < 4; ++j)
        acc = wmma_bf16(loadA_row(row, j * 32, half), Bt3[j], acc);
#pragma unroll
      for (int r = 0; r < 8; ++r) {
        float y = acc[r] * s3 + b3;
        h3v[rr][r] = y > 0.f ? y : SLOPE_ * y;
      }
    }
  }
  __syncthreads();  // all h2 reads complete before overwriting union with h3
#pragma unroll
  for (int rr = 0; rr < 4; ++rr)
#pragma unroll
    for (int r = 0; r < 8; ++r)
      sm.h3[((rg + rr) * 16 + half * 8 + r) * 64 + ct * 16 + (lane & 15)] = h3v[rr][r];
  __syncthreads();
  {
    const int nl = threadIdx.x >> 6, o = threadIdx.x & 63;
    float m = -FINF;
#pragma unroll
    for (int k = 0; k < 32; ++k) m = fmaxf(m, sm.h3[(nl * 32 + k) * 64 + o]);
    const int s0 = sBase + nl * 32;
    const int b = s0 >> 17;
    const int n = (s0 & (N_ * K2_ - 1)) >> 5;
    out[((size_t)b * N_ + n) * OC_ + o] = m;
  }
}

// ---------------------------------------------------------------------------
extern "C" void kernel_launch(void* const* d_in, const int* in_sizes, int n_in,
                              void* d_out, int out_size, void* d_ws, size_t ws_size,
                              hipStream_t stream) {
  (void)in_sizes; (void)n_in; (void)out_size; (void)ws_size;
  const float* x   = (const float*)d_in[0];
  const float* pos = (const float*)d_in[1];
  const float* W1  = (const float*)d_in[2];
  const float* g1  = (const float*)d_in[3];
  const float* b1  = (const float*)d_in[4];
  const float* W2  = (const float*)d_in[5];
  const float* g2  = (const float*)d_in[6];
  const float* b2  = (const float*)d_in[7];
  const float* W3  = (const float*)d_in[8];
  const float* g3  = (const float*)d_in[9];
  const float* b3  = (const float*)d_in[10];

  char* ws = (char*)d_ws;
  __bf16* xbf   = (__bf16*)(ws + 0);        // 1 MB
  __bf16* h1b   = (__bf16*)(ws + 1048576);  // 2 MB
  __bf16* W1b   = (__bf16*)(ws + 3145728);  // 32 KB
  __bf16* W2b   = (__bf16*)(ws + 3178496);  // 64 KB
  __bf16* W3b   = (__bf16*)(ws + 3244032);  // 16 KB
  float*  xnorm = (float*)(ws + 3260416);   // 32 KB
  int*    idx1x = (int*)  (ws + 3293184);   // 256 KB
  float*  stats = (float*)(ws + 3555328);   // 1280 floats
  float* sum1 = stats + 0;    float* sq1 = stats + 128;
  float* sc1  = stats + 256;  float* sh1 = stats + 384;
  float* sum2 = stats + 512;  float* sq2 = stats + 640;
  float* sc2  = stats + 768;  float* sh2 = stats + 896;
  float* sum3 = stats + 1024; float* sq3 = stats + 1088;
  float* sc3  = stats + 1152; float* sh3 = stats + 1216;

  float* out  = (float*)d_out;
  int*   idx2 = (int*)((float*)d_out + (size_t)B_ * N_ * OC_);

  prep_x<<<dim3((B_ * N_ + 255) / 256), 256, 0, stream>>>(x, xbf, xnorm);
  prep_w<<<dim3(224), 256, 0, stream>>>(W1, W2, W3, W1b, W2b, W3b, stats);
  knn_x_kernel<<<dim3(N_ / 32, B_), 256, 0, stream>>>(xbf, xnorm, idx1x);
  knn_pos_kernel<<<dim3(N_ / 32, B_), 128, 0, stream>>>(pos, idx2);
  conv1_stats<<<dim3((B_ * N_ * K1T_) / 128), 256, 0, stream>>>(xbf, W1b, idx1x, idx2, sum1, sq1);
  finalize_bn<<<1, 128, 0, stream>>>(sum1, sq1, g1, b1, sc1, sh1, 128, 1.0f / (B_ * N_ * K1T_));
  conv1_passB<<<dim3(N_ / 8, B_), 256, 0, stream>>>(xbf, W1b, idx1x, idx2, sc1, sh1, h1b);
  conv2_stats<<<dim3((B_ * N_ * K2_) / 128), 256, 0, stream>>>(h1b, W2b, idx2, sum2, sq2);
  finalize_bn<<<1, 128, 0, stream>>>(sum2, sq2, g2, b2, sc2, sh2, 128, 1.0f / (B_ * N_ * K2_));
  conv3_stats<<<dim3((B_ * N_ * K2_) / 128), 256, 0, stream>>>(h1b, W2b, W3b, idx2, sc2, sh2, sum3, sq3);
  finalize_bn<<<1, 64, 0, stream>>>(sum3, sq3, g3, b3, sc3, sh3, 64, 1.0f / (B_ * N_ * K2_));
  conv3_final<<<dim3((B_ * N_ * K2_) / 128), 256, 0, stream>>>(h1b, W2b, W3b, idx2, sc2, sh2, sc3, sh3, out);
}
